// CompressedSparseAttention_37271726194772
// MI455X (gfx1250) — matrix operand order repaired and checked by reading, not compiled, not run on hardware
//
#include <hip/hip_runtime.h>
#include <cmath>

// ---------------- problem constants ----------------
#define S_    4096
#define D_    2048
#define M_    4
#define NC_   1024
#define TOPK_ 512
#define NH_   16
#define C_    64
#define DC_   512
#define NG_   4
#define DG_   512
#define CI_   64
#define NHI_  4

// ---------------- WMMA plumbing (gfx1250, wave32) ----------------
typedef __attribute__((ext_vector_type(16))) __bf16 v16bf_t;
typedef __attribute__((ext_vector_type(8)))  float  v8f_t;

struct __align__(16) U4 { unsigned x, y, z, w; };

union FragBf {
  v16bf_t        v;
  U4             q[2];
  unsigned short u[16];
};

__device__ __forceinline__ unsigned short f2b(float f) {
  union { float f; unsigned u; } x; x.f = f;
  unsigned r = x.u + 0x7FFFu + ((x.u >> 16) & 1u);   // round-to-nearest-even bf16
  return (unsigned short)(r >> 16);
}

__device__ __forceinline__ float wsum(float v) {
#pragma unroll
  for (int o = 16; o > 0; o >>= 1) v += __shfl_xor(v, o, 32);
  return v;
}

// A fragment: 16x32 bf16. lane&15 = M row; lane>>4 selects K-halves.
// element e: k = kh*16 + e
__device__ __forceinline__ v16bf_t loadA16(const unsigned short* A, int lda, int m, int k0, int kh) {
  FragBf f;
  f.q[0] = *(const U4*)(A + (size_t)m * lda + k0 + kh * 16);
  f.q[1] = *(const U4*)(A + (size_t)m * lda + k0 + kh * 16 + 8);
  return f.v;
}
// B fragment: 32x16 bf16 given B^T stored N-major (ldb = K).
// lane&15 = N col; element e: k = kh*16 + e  -> 16 contiguous bf16.
__device__ __forceinline__ v16bf_t loadB16(const unsigned short* BT, int ldb, int n, int k0, int kh) {
  FragBf f;
  const U4* p = (const U4*)(BT + (size_t)n * ldb + k0 + kh * 16);
  f.q[0] = p[0]; f.q[1] = p[1];
  return f.v;
}
__device__ __forceinline__ v8f_t wmma_bf(v16bf_t a, v16bf_t b, v8f_t c) {
  return __builtin_amdgcn_wmma_f32_16x16x32_bf16(false, a, false, b, (short)0, c, false, false);
}

// ---------------- pre-pass kernels ----------------
__global__ __launch_bounds__(256) void convert_bf_kernel(const float* __restrict__ src,
                                                         unsigned short* __restrict__ dst, int n) {
  int g = blockIdx.x * 256 + threadIdx.x;
  if (g < n) dst[g] = f2b(src[g]);
}

// dst[n*K + k] = bf16(src[k*N + n])  (weight (K,N) -> B^T N-major bf16)
__global__ __launch_bounds__(256) void transpose_bf_kernel(const float* __restrict__ src,
                                                           unsigned short* __restrict__ dst,
                                                           int K, int N) {
  int g = blockIdx.x * 256 + threadIdx.x;
  if (g >= K * N) return;
  int k = g / N, n = g - k * N;
  dst[(size_t)n * K + k] = f2b(src[g]);
}

// ---------------- generic bf16 WMMA GEMM, 16x64 strip per wave ----------------
// C[M,N] = A[M,K] (bf16, lda) * B (via BT[N,K] bf16, ldb). Optional f32 and bf16 outputs.
// Ntiles must be a multiple of 4 (true for every call site here).
__global__ __launch_bounds__(256)
void gemm_bf16_kernel(const unsigned short* __restrict__ A, int lda,
                      const unsigned short* __restrict__ BT, int ldb,
                      float* __restrict__ Cf, unsigned short* __restrict__ Cb,
                      int ldc, int Mtiles, int Ntiles, int K) {
  int nquads = Ntiles >> 2;
  int wv = blockIdx.x * 8 + (threadIdx.x >> 5);
  if (wv >= Mtiles * nquads) return;                 // wave-uniform
  int lane = threadIdx.x & 31;
  int kh = lane >> 4, n16 = lane & 15;
  int mt = wv / nquads, nq = wv - mt * nquads;
  int m0 = mt * 16, n0 = nq * 64;
  int m = m0 + n16;
  v8f_t acc0 = {}, acc1 = {}, acc2 = {}, acc3 = {};
  const unsigned short* brow0 = BT + (size_t)(n0 + n16) * ldb;
  for (int k0 = 0; k0 < K; k0 += 32) {
    v16bf_t a = loadA16(A, lda, m, k0, kh);          // one A frag reused 4x
    // prefetch next K-block of the streams (global_prefetch_b8)
    __builtin_prefetch(A + (size_t)m * lda + k0 + 256, 0, 1);
    __builtin_prefetch(brow0 + k0 + 256, 0, 1);
    v16bf_t b0 = loadB16(BT, ldb, n0 + n16,      k0, kh);
    v16bf_t b1 = loadB16(BT, ldb, n0 + 16 + n16, k0, kh);
    v16bf_t b2 = loadB16(BT, ldb, n0 + 32 + n16, k0, kh);
    v16bf_t b3 = loadB16(BT, ldb, n0 + 48 + n16, k0, kh);
    acc0 = wmma_bf(a, b0, acc0);
    acc1 = wmma_bf(a, b1, acc1);
    acc2 = wmma_bf(a, b2, acc2);
    acc3 = wmma_bf(a, b3, acc3);
  }
  int row0 = m0 + kh * 8;
#pragma unroll
  for (int r = 0; r < 8; ++r) {
    float v0 = acc0[r], v1 = acc1[r], v2 = acc2[r], v3 = acc3[r];
    size_t o = (size_t)(row0 + r) * ldc + n0 + n16;
    if (Cf) { Cf[o] = v0; Cf[o + 16] = v1; Cf[o + 32] = v2; Cf[o + 48] = v3; }
    if (Cb) { Cb[o] = f2b(v0); Cb[o + 16] = f2b(v1); Cb[o + 32] = f2b(v2); Cb[o + 48] = f2b(v3); }
  }
}

// ---------------- small VALU kernels ----------------
__global__ __launch_bounds__(256) void hw_kernel(const float* __restrict__ x,
                                                 const float* __restrict__ ww,
                                                 float* __restrict__ hw) {
  int s = blockIdx.x * 8 + (threadIdx.x >> 5);
  if (s >= S_) return;
  int lane = threadIdx.x & 31;
  float a0 = 0, a1 = 0, a2 = 0, a3 = 0;
  const float* xr = x + (size_t)s * D_;
  for (int k = lane; k < D_; k += 32) {
    float xv = xr[k];
    const float* w4 = ww + k * 4;
    a0 += xv * w4[0]; a1 += xv * w4[1]; a2 += xv * w4[2]; a3 += xv * w4[3];
  }
  a0 = wsum(a0); a1 = wsum(a1); a2 = wsum(a2); a3 = wsum(a3);
  if (lane == 0) { float* o = hw + s * 4; o[0] = a0; o[1] = a1; o[2] = a2; o[3] = a3; }
}

__global__ __launch_bounds__(256) void korig_kernel(const float* __restrict__ x,
                                                    unsigned short* __restrict__ kb) {
  int g = blockIdx.x * 256 + threadIdx.x;
  if (g >= NC_ * D_) return;
  int n = g >> 11, d = g & (D_ - 1);
  const float* b = x + (size_t)(n * 4) * D_ + d;
  kb[g] = f2b(0.25f * (b[0] + b[D_] + b[2 * D_] + b[3 * D_]));
}

// softmax over 8 gate logits -> compressed KV (NC x 64 f32)
__global__ __launch_bounds__(256) void compress_kernel(const float* __restrict__ pk,  // S x 256
                                                       const float* __restrict__ b_a,
                                                       const float* __restrict__ b_b,
                                                       float* __restrict__ comp) {
  int g = blockIdx.x * 256 + threadIdx.x;
  if (g >= NC_ * 64) return;
  int n = g >> 6, c = g & 63;
  float lg[8], vl[8];
#pragma unroll
  for (int m = 0; m < 4; ++m) {
    if (n == 0) { lg[m] = -1e30f + b_b[m * 64 + c]; vl[m] = 0.f; }
    else {
      size_t rp = (size_t)((n - 1) * 4 + m) * 256;
      lg[m] = pk[rp + 192 + c] + b_b[m * 64 + c];   // prev z_b + b_b
      vl[m] = pk[rp + 64 + c];                      // prev c_b
    }
    size_t rc = (size_t)(n * 4 + m) * 256;
    lg[4 + m] = pk[rc + 128 + c] + b_a[m * 64 + c]; // z_a + b_a
    vl[4 + m] = pk[rc + c];                         // c_a
  }
  float mx = lg[0];
#pragma unroll
  for (int i = 1; i < 8; ++i) mx = fmaxf(mx, lg[i]);
  float se = 0.f, out = 0.f;
#pragma unroll
  for (int i = 0; i < 8; ++i) { float e = __expf(lg[i] - mx); se += e; out += e * vl[i]; }
  comp[g] = out / se;
}

__global__ __launch_bounds__(256) void ln_kc_kernel(const float* __restrict__ comp,
                                                    const float* __restrict__ w,
                                                    const float* __restrict__ b,
                                                    unsigned short* __restrict__ kcbf) {
  int n = blockIdx.x * 8 + (threadIdx.x >> 5);
  if (n >= NC_) return;
  int lane = threadIdx.x & 31;
  const float* row = comp + n * 64;
  float x0 = row[lane], x1 = row[lane + 32];
  float mu = wsum(x0 + x1) * (1.f / 64.f);
  float d0 = x0 - mu, d1 = x1 - mu;
  float var = wsum(d0 * d0 + d1 * d1) * (1.f / 64.f);
  float rs = rsqrtf(var + 1e-6f);
  kcbf[n * 64 + lane]      = f2b(d0 * rs * w[lane] + b[lane]);
  kcbf[n * 64 + lane + 32] = f2b(d1 * rs * w[lane + 32] + b[lane + 32]);
}

// LN + RoPE on q rows; one wave per (s, head)
__global__ __launch_bounds__(256) void q_ln_rope_kernel(const float* __restrict__ qraw,
                                                        const float* __restrict__ qw,
                                                        const float* __restrict__ qb,
                                                        unsigned short* __restrict__ qbf) {
  int gw = blockIdx.x * 8 + (threadIdx.x >> 5);
  int s = gw >> 4, h = gw & 15;
  int lane = threadIdx.x & 31;
  int c0 = lane * 2;
  const float* row = qraw + (size_t)s * 1024 + h * 64;
  float x0 = row[c0], x1 = row[c0 + 1];
  float mu = wsum(x0 + x1) * (1.f / 64.f);
  float d0 = x0 - mu, d1 = x1 - mu;
  float var = wsum(d0 * d0 + d1 * d1) * (1.f / 64.f);
  float rs = rsqrtf(var + 1e-6f);
  float y0 = d0 * rs * qw[c0] + qb[c0];
  float y1 = d1 * rs * qw[c0 + 1] + qb[c0 + 1];
  float o0 = y0, o1 = y1;
  if (c0 >= 32) {                                   // rope on last 32 dims
    int j = (c0 - 32) >> 1;
    float inv = __expf(-(float)j * 0.57564627324851148f);  // 10000^(-j/16)
    float ang = (float)s * inv;
    float sn, cs; __sincosf(ang, &sn, &cs);
    o0 = y0 * cs - y1 * sn;
    o1 = y0 * sn + y1 * cs;
  }
  size_t o = (size_t)s * 1024 + h * 64 + c0;
  qbf[o] = f2b(o0); qbf[o + 1] = f2b(o1);
}

// indexer scores: relu(q_i . k_proj) . hw, causal mask — WMMA with fused epilogue
__global__ __launch_bounds__(256)
void iscores_kernel(const unsigned short* __restrict__ Qi,  // (S*4) x 64 bf16
                    const unsigned short* __restrict__ Kp,  // NC x 64 bf16 (= B^T)
                    const float* __restrict__ hw,           // S x 4
                    float* __restrict__ isc) {              // S x NC
  const int Ntiles = NC_ / 16;
  int wv = blockIdx.x * 8 + (threadIdx.x >> 5);
  if (wv >= (S_ * NHI_ / 16) * Ntiles) return;
  int lane = threadIdx.x & 31;
  int kh = lane >> 4, n16 = lane & 15;
  int mt = wv / Ntiles, nt = wv - mt * Ntiles;
  int m0 = mt * 16, n0 = nt * 16;
  v8f_t acc = {};
  for (int k0 = 0; k0 < 64; k0 += 32) {
    v16bf_t a = loadA16(Qi, 64, m0 + n16, k0, kh);
    v16bf_t b = loadB16(Kp, 64, n0 + n16, k0, kh);
    acc = wmma_bf(a, b, acc);
  }
  int n = n0 + n16;
  int sb = (m0 + kh * 8) >> 2;                       // rows cover s = sb, sb+1 (4 heads each)
  float v0 = 0.f, v1 = 0.f;
#pragma unroll
  for (int r = 0; r < 4; ++r) { float d = acc[r];     if (d > 0.f) v0 += d * hw[sb * 4 + r]; }
#pragma unroll
  for (int r = 4; r < 8; ++r) { float d = acc[r];     if (d > 0.f) v1 += d * hw[(sb + 1) * 4 + (r - 4)]; }
  isc[(size_t)sb * NC_ + n]       = (n < sb)     ? v0 : -INFINITY;
  isc[(size_t)(sb + 1) * NC_ + n] = (n < sb + 1) ? v1 : -INFINITY;
}

// exact top-512-of-1024: bitonic sort (desc value, asc index tie-break = jax.lax.top_k)
__global__ __launch_bounds__(512) void topk_kernel(const float* __restrict__ isc,
                                                   int* __restrict__ topidx) {
  __shared__ float v[1024];
  __shared__ int   id[1024];
  int s = blockIdx.x, t = threadIdx.x;
  for (int i = t; i < 1024; i += 512) { v[i] = isc[(size_t)s * NC_ + i]; id[i] = i; }
  __syncthreads();
  for (int k = 2; k <= 1024; k <<= 1) {
    for (int j = k >> 1; j > 0; j >>= 1) {
      for (int i = t; i < 1024; i += 512) {
        int l = i ^ j;
        if (l > i) {
          float va = v[i], vb = v[l];
          int ia = id[i], ib = id[l];
          bool aFirst = (va > vb) || (va == vb && ia < ib);
          bool desc = ((i & k) == 0);
          if (desc != aFirst) { v[i] = vb; v[l] = va; id[i] = ib; id[l] = ia; }
        }
      }
      __syncthreads();
    }
  }
  topidx[(size_t)s * TOPK_ + t] = id[t];
}

// gathered attention: scores = q.k_sel/8 -> mask/exp -> sink softmax -> P.k_sel
__global__ __launch_bounds__(256)
void attention_kernel(const unsigned short* __restrict__ qbf,   // S x 1024 bf16
                      const unsigned short* __restrict__ kcbf,  // NC x 64 bf16
                      const int* __restrict__ topidx,           // S x 512
                      const float* __restrict__ sink,           // 16
                      unsigned short* __restrict__ outbf) {     // S x 1024 bf16
  __shared__ int idx[TOPK_];
  __shared__ __align__(16) float p[16 * TOPK_];
  __shared__ __align__(16) unsigned short pb[16 * TOPK_];
  __shared__ float denom[16];
  int s = blockIdx.x, tid = threadIdx.x;
  for (int i = tid; i < TOPK_; i += 256) idx[i] = topidx[(size_t)s * TOPK_ + i];
  __syncthreads();
  int lane = tid & 31, wv = tid >> 5;
  int kh = lane >> 4, n16 = lane & 15;
  const unsigned short* Aq = qbf + (size_t)s * 1024;

  // phase 1: scores (16 heads x 512 keys), gather-B through top_idx
  for (int t = 0; t < 4; ++t) {
    int n0 = wv * 64 + t * 16;
    int n = n0 + n16;
    int chunk = idx[n];
    v8f_t acc = {};
    for (int k0 = 0; k0 < 64; k0 += 32) {
      v16bf_t a = loadA16(Aq, 64, n16, k0, kh);
      v16bf_t b = loadB16(kcbf, 64, chunk, k0, kh);
      acc = wmma_bf(a, b, acc);
    }
    bool keep = (s < chunk * M_);
#pragma unroll
    for (int r = 0; r < 8; ++r) {
      int head = kh * 8 + r;
      float e = keep ? __expf(acc[r] * 0.125f) : 0.0f;
      p[head * TOPK_ + n] = e;
      pb[head * TOPK_ + n] = f2b(e);
    }
  }
  __syncthreads();
  if (tid < 16) {
    float d = 0.f;
    for (int n = 0; n < TOPK_; ++n) d += p[tid * TOPK_ + n];
    denom[tid] = d + __expf(sink[tid]);
  }
  __syncthreads();
  // phase 2: out = P @ k_sel (16 x 64), waves 0..3, strided gather-B
  if (wv < 4) {
    int c = wv * 16 + n16;
    v8f_t acc = {};
    for (int k0 = 0; k0 < TOPK_; k0 += 32) {
      v16bf_t a = loadA16(pb, TOPK_, n16, k0, kh);
      FragBf fb;
#pragma unroll
      for (int e = 0; e < 16; ++e) {
        int k = k0 + kh * 16 + e;
        fb.u[e] = kcbf[(size_t)idx[k] * 64 + c];
      }
      acc = wmma_bf(a, fb.v, acc);
    }
#pragma unroll
    for (int r = 0; r < 8; ++r) {
      int head = kh * 8 + r;
      outbf[(size_t)s * 1024 + head * 64 + c] = f2b(acc[r] / denom[head]);
    }
  }
}

// ---------------- launcher ----------------
extern "C" void kernel_launch(void* const* d_in, const int* in_sizes, int n_in,
                              void* d_out, int out_size, void* d_ws, size_t ws_size,
                              hipStream_t stream) {
  (void)in_sizes; (void)n_in; (void)out_size; (void)ws_size;
  const float* x      = (const float*)d_in[0];
  const float* w_kv_a = (const float*)d_in[1];
  const float* w_kv_b = (const float*)d_in[2];
  const float* w_z_a  = (const float*)d_in[3];
  const float* w_z_b  = (const float*)d_in[4];
  const float* b_a    = (const float*)d_in[5];
  const float* b_b    = (const float*)d_in[6];
  const float* w_dq   = (const float*)d_in[7];
  const float* w_iuq  = (const float*)d_in[8];
  const float* w_w    = (const float*)d_in[9];
  const float* w_k    = (const float*)d_in[10];
  const float* w_uq   = (const float*)d_in[11];
  const float* o_down = (const float*)d_in[12];
  const float* o_up   = (const float*)d_in[13];
  const float* kvn_w  = (const float*)d_in[14];
  const float* kvn_b  = (const float*)d_in[15];
  const float* qn_w   = (const float*)d_in[16];
  const float* qn_b   = (const float*)d_in[17];
  const float* sink   = (const float*)d_in[18];

  char* base = (char*)d_ws;
  size_t off = 0;
  auto alloc = [&](size_t bytes) -> void* {
    void* p = base + off;
    off += (bytes + 255) & ~(size_t)255;
    return p;
  };
  typedef unsigned short u16;
  u16*   x_bf     = (u16*)alloc((size_t)S_ * D_ * 2);
  u16*   wkvT     = (u16*)alloc((size_t)256 * D_ * 2);
  u16*   wdqT     = (u16*)alloc((size_t)DC_ * D_ * 2);
  u16*   wiuqT    = (u16*)alloc((size_t)256 * DC_ * 2);
  u16*   wuqT     = (u16*)alloc((size_t)1024 * DC_ * 2);
  u16*   wkT      = (u16*)alloc((size_t)CI_ * D_ * 2);
  u16*   odT      = (u16*)alloc((size_t)NG_ * DG_ * 256 * 2);
  u16*   oupT     = (u16*)alloc((size_t)2048 * 2048 * 2);
  float* projkv   = (float*)alloc((size_t)S_ * 256 * 4);
  u16*   cq_bf    = (u16*)alloc((size_t)S_ * DC_ * 2);
  u16*   qi_bf    = (u16*)alloc((size_t)S_ * 256 * 2);
  float* qraw     = (float*)alloc((size_t)S_ * 1024 * 4);
  float* hwbuf    = (float*)alloc((size_t)S_ * 4 * 4);
  u16*   korig_bf = (u16*)alloc((size_t)NC_ * D_ * 2);
  u16*   kproj_bf = (u16*)alloc((size_t)NC_ * CI_ * 2);
  float* comp     = (float*)alloc((size_t)NC_ * 64 * 4);
  u16*   kc_bf    = (u16*)alloc((size_t)NC_ * 64 * 2);
  float* isc      = (float*)alloc((size_t)S_ * NC_ * 4);
  int*   topidx   = (int*)alloc((size_t)S_ * TOPK_ * 4);
  u16*   q_bf     = (u16*)alloc((size_t)S_ * 1024 * 2);
  u16*   attn_bf  = (u16*)alloc((size_t)S_ * 1024 * 2);
  u16*   g_bf     = (u16*)alloc((size_t)S_ * 2048 * 2);

  auto cvt = [&](const float* s, u16* d, int n) {
    convert_bf_kernel<<<(n + 255) / 256, 256, 0, stream>>>(s, d, n);
  };
  auto trp = [&](const float* s, u16* d, int K, int N) {
    transpose_bf_kernel<<<(K * N + 255) / 256, 256, 0, stream>>>(s, d, K, N);
  };
  auto gemm = [&](const u16* A, int lda, const u16* BT, int ldb,
                  float* Cf, u16* Cb, int ldc, int Mt, int Nt, int K) {
    int waves = Mt * (Nt >> 2);                      // 16x64 strip per wave
    gemm_bf16_kernel<<<(waves + 7) / 8, 256, 0, stream>>>(A, lda, BT, ldb, Cf, Cb, ldc, Mt, Nt, K);
  };

  // 1. precision conversion / weight transposes
  cvt(x, x_bf, S_ * D_);
  trp(w_kv_a, wkvT,            D_, 64);
  trp(w_kv_b, wkvT + 64 * D_,  D_, 64);
  trp(w_z_a,  wkvT + 128 * D_, D_, 64);
  trp(w_z_b,  wkvT + 192 * D_, D_, 64);
  trp(w_dq,  wdqT,  D_, DC_);
  trp(w_iuq, wiuqT, DC_, 256);
  trp(w_uq,  wuqT,  DC_, 1024);
  trp(w_k,   wkT,   D_, CI_);
  for (int gr = 0; gr < NG_; ++gr)
    trp(o_down + (size_t)gr * 256 * DG_, odT + (size_t)gr * DG_ * 256, 256, DG_);
  trp(o_up, oupT, 2048, 2048);

  // 2. projections (WMMA GEMMs)
  gemm(x_bf, D_, wkvT, D_, projkv, nullptr, 256, S_ / 16, 16, D_);        // c_a|c_b|z_a|z_b
  gemm(x_bf, D_, wdqT, D_, nullptr, cq_bf, DC_, S_ / 16, DC_ / 16, D_);   // c_q
  gemm(cq_bf, DC_, wiuqT, DC_, nullptr, qi_bf, 256, S_ / 16, 16, DC_);    // q_i
  gemm(cq_bf, DC_, wuqT, DC_, qraw, nullptr, 1024, S_ / 16, 64, DC_);     // q (pre-LN)
  hw_kernel<<<S_ / 8, 256, 0, stream>>>(x, w_w, hwbuf);
  korig_kernel<<<(NC_ * D_ + 255) / 256, 256, 0, stream>>>(x, korig_bf);
  gemm(korig_bf, D_, wkT, D_, nullptr, kproj_bf, CI_, NC_ / 16, CI_ / 16, D_); // k_proj

  // 3. gate softmax -> compressed KV -> LN
  compress_kernel<<<(NC_ * 64 + 255) / 256, 256, 0, stream>>>(projkv, b_a, b_b, comp);
  ln_kc_kernel<<<NC_ / 8, 256, 0, stream>>>(comp, kvn_w, kvn_b, kc_bf);

  // 4. indexer scores + exact top-k
  iscores_kernel<<<(S_ * NHI_ / 16) * (NC_ / 16) / 8, 256, 0, stream>>>(qi_bf, kproj_bf, hwbuf, isc);
  topk_kernel<<<S_, 512, 0, stream>>>(isc, topidx);

  // 5. q LN + rope, then gathered attention
  q_ln_rope_kernel<<<S_ * NH_ / 8, 256, 0, stream>>>(qraw, qn_w, qn_b, q_bf);
  attention_kernel<<<S_, 256, 0, stream>>>(q_bf, kc_bf, topidx, sink, attn_bf);

  // 6. output projection: block-diag o_down then dense o_up -> d_out (f32)
  for (int gr = 0; gr < NG_; ++gr)
    gemm(attn_bf + gr * 256, 1024, odT + (size_t)gr * DG_ * 256, 256,
         nullptr, g_bf + gr * DG_, 2048, S_ / 16, DG_ / 16, 256);
  gemm(g_bf, 2048, oupT, 2048, (float*)d_out, nullptr, 2048, S_ / 16, 2048 / 16, 2048);
}
